// TransformerEncoder_77567109366074
// MI455X (gfx1250) — matrix operand order repaired and checked by reading, not compiled, not run on hardware
//
#include <hip/hip_runtime.h>

#define B_   32
#define L_   512
#define D_   1024
#define DFF_ 4096
#define H_   16
#define NL_  4
#define DH_  64
#define BL_  (B_ * L_)

typedef __attribute__((ext_vector_type(16))) __bf16 v16bf;
typedef __attribute__((ext_vector_type(8)))  float  v8f;
typedef __attribute__((ext_vector_type(4)))  unsigned int v4u;
typedef __attribute__((ext_vector_type(8)))  int  v8i;
typedef __attribute__((ext_vector_type(4)))  int  v4i;

union FragBF { v16bf v; unsigned int u[8]; };
union AccF   { v8f   v; float        f[8]; };

#if __has_builtin(__builtin_amdgcn_tensor_load_to_lds) && __has_builtin(__builtin_amdgcn_s_wait_tensorcnt)
#define USE_TDM 1
#else
#define USE_TDM 0
#endif

__device__ __forceinline__ unsigned short f2bf(float x) {
  union { float f; unsigned int u; } c; c.f = x;
  unsigned int u = c.u;
  unsigned int r = u + 0x7fffu + ((u >> 16) & 1u);   // round-to-nearest-even
  return (unsigned short)(r >> 16);
}
__device__ __forceinline__ unsigned int pack2bf(float lo, float hi) {
  return (unsigned int)f2bf(lo) | ((unsigned int)f2bf(hi) << 16);
}
__device__ __forceinline__ float gelu_tanh(float x) {
  const float c = 0.7978845608028654f;
  return 0.5f * x * (1.0f + tanhf(c * (x + 0.044715f * x * x * x)));
}
__device__ __forceinline__ float wred_sum(float v) {
#pragma unroll
  for (int o = 16; o >= 1; o >>= 1) v += __shfl_xor(v, o, 32);
  return v;
}
__device__ __forceinline__ float wred_max(float v) {
#pragma unroll
  for (int o = 16; o >= 1; o >>= 1) v = fmaxf(v, __shfl_xor(v, o, 32));
  return v;
}

// ---------------------------------------------------------------------------
// fp32 -> bf16 conversion (weights)
// ---------------------------------------------------------------------------
__global__ void __launch_bounds__(256)
cvt_bf16_kernel(const float* __restrict__ src, unsigned short* __restrict__ dst, size_t n) {
  size_t i = (size_t)blockIdx.x * 256 + threadIdx.x;
  size_t stride = (size_t)gridDim.x * 256;
  for (; i < n; i += stride) dst[i] = f2bf(src[i]);
}

// ---------------------------------------------------------------------------
// x + positional-encoding + speaker embedding; fp32 and/or bf16 outputs
// ---------------------------------------------------------------------------
__global__ void __launch_bounds__(256)
add_pe_kernel(const float* __restrict__ x, const float* __restrict__ spk,
              float* __restrict__ outF, unsigned short* __restrict__ outB) {
  size_t i = (size_t)blockIdx.x * 256 + threadIdx.x;
  const size_t total = (size_t)BL_ * D_;
  if (i >= total) return;
  int d = (int)(i % D_);
  size_t r = i / D_;
  int l  = (int)(r % L_);
  int bb = (int)(r / L_);
  int hi = d >> 1;
  float f = __expf((-2.0f * (float)hi) * (9.210340371976184f / (float)D_)); // ln(10000)/D
  float ang = (float)l * f;
  float pe = (d & 1) ? __cosf(ang) : __sinf(ang);
  float v = x[i] + pe + spk[(size_t)bb * D_ + d];
  if (outF) outF[i] = v;
  if (outB) outB[i] = f2bf(v);
}

// ---------------------------------------------------------------------------
// LayerNorm over D=1024, one block (256 thr) per row; fp32 and/or bf16 out
// ---------------------------------------------------------------------------
__global__ void __launch_bounds__(256)
layernorm_kernel(const float* __restrict__ x, const float* __restrict__ g,
                 const float* __restrict__ b, float* __restrict__ outF,
                 unsigned short* __restrict__ outB) {
  __shared__ float sbuf[8];
  __shared__ float bcast;
  const int tid = threadIdx.x;
  const int lane = tid & 31, wave = tid >> 5;
  const size_t row = blockIdx.x;
  const float* xr = x + row * D_;
  float v[4];
  float s = 0.0f;
#pragma unroll
  for (int i = 0; i < 4; ++i) { v[i] = xr[tid + i * 256]; s += v[i]; }
  s = wred_sum(s);
  if (lane == 0) sbuf[wave] = s;
  __syncthreads();
  if (tid == 0) { float t = 0; for (int i = 0; i < 8; ++i) t += sbuf[i]; bcast = t; }
  __syncthreads();
  float mu = bcast * (1.0f / (float)D_);
  float s2 = 0.0f;
#pragma unroll
  for (int i = 0; i < 4; ++i) { float d = v[i] - mu; s2 += d * d; }
  s2 = wred_sum(s2);
  __syncthreads();
  if (lane == 0) sbuf[wave] = s2;
  __syncthreads();
  if (tid == 0) { float t = 0; for (int i = 0; i < 8; ++i) t += sbuf[i]; bcast = t; }
  __syncthreads();
  float rstd = rsqrtf(bcast * (1.0f / (float)D_) + 1e-6f);
#pragma unroll
  for (int i = 0; i < 4; ++i) {
    int c = tid + i * 256;
    float y = (v[i] - mu) * rstd * g[c] + b[c];
    size_t o = row * D_ + c;
    if (outF) outF[o] = y;
    if (outB) outB[o] = f2bf(y);
  }
}

// ---------------------------------------------------------------------------
// TDM helper: 2D tile load (128 rows x 32 bf16) into padded LDS (stride 34)
//   tensor: [M, K] bf16, row-major; pad 1 DWORD after every 16 DWORDs (1 row)
// ---------------------------------------------------------------------------
#if USE_TDM
__device__ __forceinline__ void tdm_load_a_tile(const unsigned short* gsrc,
                                                unsigned ldsOff, int M, int K) {
  unsigned long long ga = (unsigned long long)gsrc;
  unsigned galo = (unsigned)__builtin_amdgcn_readfirstlane((int)(unsigned)(ga & 0xffffffffull));
  unsigned gahi = (unsigned)__builtin_amdgcn_readfirstlane((int)(unsigned)(ga >> 32));
  unsigned lds  = (unsigned)__builtin_amdgcn_readfirstlane((int)ldsOff);
  unsigned uK = (unsigned)K, uM = (unsigned)M;
  v4u g0 = { 1u,                                   // count=1, user desc
             lds,                                  // lds_addr
             galo,                                 // global_addr[31:0]
             (gahi & 0x1ffffffu) | (2u << 30) };   // global_addr[56:32] | type=2
  v8i g1 = {
    (int)((1u << 16) | (1u << 20) | (3u << 22)),   // data_size=2B, pad_en, ivl=16DW, amt=1DW
    (int)((uK & 0xffffu) << 16),                   // tensor_dim0 lo16
    (int)((uK >> 16) | ((uM & 0xffffu) << 16)),    // tensor_dim0 hi16 | tensor_dim1 lo16
    (int)((uM >> 16) | (32u << 16)),               // tensor_dim1 hi16 | tile_dim0=32
    (int)128,                                      // tile_dim1=128, tile_dim2=0
    (int)uK,                                       // tensor_dim0_stride lo32
    0, 0 };                                        // stride hi, dim1_stride=0
  v4i gz4 = { 0, 0, 0, 0 };
  v8i gz8 = { 0, 0, 0, 0, 0, 0, 0, 0 };
  __builtin_amdgcn_tensor_load_to_lds(g0, g1, gz4, gz4, gz8, 0);
}
#endif

// ---------------------------------------------------------------------------
// GEMM staging helpers
// ---------------------------------------------------------------------------
#define LDT 34

__device__ __forceinline__ void stage_a_vgpr(const unsigned short* __restrict__ A,
                                             unsigned short* dstAs, int bm, int K,
                                             int k0, int tid) {
#pragma unroll
  for (int it = 0; it < 2; ++it) {
    int idx = tid + it * 256;            // 0..511
    int row = idx >> 2;
    int q   = (idx & 3) << 3;            // bf16 col 0/8/16/24
    const uint4 v = *(const uint4*)(A + (size_t)(bm + row) * K + k0 + q);
    *(unsigned int*)&dstAs[row * LDT + q + 0] = v.x;
    *(unsigned int*)&dstAs[row * LDT + q + 2] = v.y;
    *(unsigned int*)&dstAs[row * LDT + q + 4] = v.z;
    *(unsigned int*)&dstAs[row * LDT + q + 6] = v.w;
  }
}
__device__ __forceinline__ void load_w_regs(const unsigned short* __restrict__ W,
                                            int N, int k0, int bn, int tid, uint4* r) {
#pragma unroll
  for (int it = 0; it < 2; ++it) {
    int idx = tid + it * 256;            // 0..511
    int rr = idx >> 4;                   // k row 0..31
    int cc = (idx & 15) << 3;            // n col 0..120
    r[it] = *(const uint4*)(W + (size_t)(k0 + rr) * N + bn + cc);
  }
}
__device__ __forceinline__ void store_w_lds(unsigned short* Bsb, int tid, const uint4* r) {
#pragma unroll
  for (int it = 0; it < 2; ++it) {
    int idx = tid + it * 256;
    int rr = idx >> 4;
    int cc = (idx & 15) << 3;
    unsigned int ww[4] = { r[it].x, r[it].y, r[it].z, r[it].w };
#pragma unroll
    for (int p = 0; p < 4; ++p) {
      Bsb[(cc + 2 * p + 0) * LDT + rr] = (unsigned short)(ww[p] & 0xffffu);
      Bsb[(cc + 2 * p + 1) * LDT + rr] = (unsigned short)(ww[p] >> 16);
    }
  }
}

// ---------------------------------------------------------------------------
// bf16 WMMA GEMM: C = A[M,K] * W[K,N]; epilogue: (+bias)*scale, +res, gelu,
// fp32/bf16 stores. 128x128 block tile, K-step 32, double-buffered LDS,
// A tile moved by the Tensor Data Mover (TDM), W tile VGPR-transposed.
// ---------------------------------------------------------------------------
__global__ void __launch_bounds__(256)
gemm_bf16_wmma(const unsigned short* __restrict__ A,
               const unsigned short* __restrict__ W,
               const float* __restrict__ bias,
               const float* __restrict__ res,
               float* __restrict__ outF,
               unsigned short* __restrict__ outB,
               int M, int N, int K, float scale, int do_gelu) {
  __shared__ unsigned short As[2][128 * LDT];
  __shared__ unsigned short Bs[2][128 * LDT];

  const int tid  = threadIdx.x;
  const int lane = tid & 31;
  const int wave = tid >> 5;
  const int wr = wave >> 2;            // 0..1 -> 64 rows
  const int wc = wave & 3;             // 0..3 -> 32 cols
  const int l16  = lane & 15;
  const int halo = (lane >> 4) << 3;   // 0 or 8  (C row-offset)
  const int kb16 = (lane >> 4) << 4;   // 0 or 16 (B frag K-offset)
  const int bm = blockIdx.y * 128;
  const int bn = blockIdx.x * 128;

  AccF acc[4][2];
#pragma unroll
  for (int mt = 0; mt < 4; ++mt)
#pragma unroll
    for (int nt = 0; nt < 2; ++nt)
#pragma unroll
      for (int i = 0; i < 8; ++i) acc[mt][nt].f[i] = 0.0f;

  // ---- prologue: stage K-tile 0 into buffer 0 ----
#if USE_TDM
  if (wave == 0)
    tdm_load_a_tile(A + (size_t)bm * K,
                    (unsigned)(unsigned long long)&As[0][0], M, K);
#else
  stage_a_vgpr(A, &As[0][0], bm, K, 0, tid);
#endif
  {
    uint4 wrg[2];
    load_w_regs(W, N, 0, bn, tid, wrg);
    store_w_lds(&Bs[0][0], tid, wrg);
  }
#if USE_TDM
  if (wave == 0) __builtin_amdgcn_s_wait_tensorcnt(0);
#endif
  __syncthreads();

  int cur = 0;
  for (int k0 = 0; k0 < K; k0 += 32, cur ^= 1) {
    const int k1 = k0 + 32;
    const bool has_next = (k1 < K);
    uint4 wrg[2];
    if (has_next) {
#if USE_TDM
      if (wave == 0)
        tdm_load_a_tile(A + (size_t)bm * K + k1,
                        (unsigned)(unsigned long long)&As[cur ^ 1][0], M, K);
#else
      stage_a_vgpr(A, &As[cur ^ 1][0], bm, K, k1, tid);
#endif
      load_w_regs(W, N, k1, bn, tid, wrg);   // in flight during WMMA below
    }

    // ---- fragments + 8x WMMA from current buffers ----
    const unsigned short* Asc = &As[cur][0];
    const unsigned short* Bsc = &Bs[cur][0];
    FragBF af[4], bf[2];
#pragma unroll
    for (int mt = 0; mt < 4; ++mt) {
      int base = (wr * 64 + mt * 16 + l16) * LDT;
#pragma unroll
      for (int j = 0; j < 8; ++j) {
        af[mt].u[j]     = *(const unsigned int*)&Asc[base + kb16 + 2 * j];
      }
    }
#pragma unroll
    for (int nt = 0; nt < 2; ++nt) {
      int base = (wc * 32 + nt * 16 + l16) * LDT + kb16;
#pragma unroll
      for (int j = 0; j < 8; ++j)
        bf[nt].u[j] = *(const unsigned int*)&Bsc[base + 2 * j];
    }
#pragma unroll
    for (int mt = 0; mt < 4; ++mt)
#pragma unroll
      for (int nt = 0; nt < 2; ++nt)
        acc[mt][nt].v = __builtin_amdgcn_wmma_f32_16x16x32_bf16(
            false, af[mt].v, false, bf[nt].v, (short)0, acc[mt][nt].v, false, false);

    if (has_next) {
      store_w_lds(&Bs[cur ^ 1][0], tid, wrg);
#if USE_TDM
      if (wave == 0) __builtin_amdgcn_s_wait_tensorcnt(0);
#endif
    }
    __syncthreads();
  }

  // ---- epilogue ----
#pragma unroll
  for (int nt = 0; nt < 2; ++nt) {
    int gn = bn + wc * 32 + nt * 16 + l16;
    float bv = bias[gn];
#pragma unroll
    for (int mt = 0; mt < 4; ++mt) {
#pragma unroll
      for (int i = 0; i < 8; ++i) {
        int gm = bm + wr * 64 + mt * 16 + halo + i;
        float v = (acc[mt][nt].f[i] + bv) * scale;
        size_t o = (size_t)gm * N + gn;
        if (res)     v += res[o];
        if (do_gelu) v  = gelu_tanh(v);
        if (outF) outF[o] = v;
        if (outB) outB[o] = f2bf(v);
      }
    }
  }
}

// ---------------------------------------------------------------------------
// Attention: one block per (b, h, 64-query tile). 4 waves, WMMA QK^T + PV,
// full scores row in LDS (64x512 fp32) + V^T staged in LDS.
// ---------------------------------------------------------------------------
#define LSC 512
#define LVT 520

__global__ void __launch_bounds__(128)
attention_wmma(const unsigned short* __restrict__ Q,
               const unsigned short* __restrict__ K,
               const unsigned short* __restrict__ V,
               unsigned short* __restrict__ O) {
  extern __shared__ char smem[];
  float* sc = (float*)smem;                                       // 64 x 512 f32
  unsigned short* Vt = (unsigned short*)(smem + 64 * LSC * 4);    // 64 x LVT bf16

  const int tid  = threadIdx.x;
  const int lane = tid & 31;
  const int wave = tid >> 5;           // 0..3
  const int l16  = lane & 15;
  const int halo = (lane >> 4) << 3;
  const int kb16 = (lane >> 4) << 4;
  const int qt = blockIdx.x;
  const int h  = blockIdx.y;
  const int b  = blockIdx.z;

  const size_t base = (size_t)b * L_ * D_ + (size_t)h * DH_;
  const unsigned short* Qp = Q + base;
  const unsigned short* Kp = K + base;
  const unsigned short* Vp = V + base;
  unsigned short* Op = O + base;

  // ---- stage V^T: Vt[dh][kpos] ----
  for (int it = 0; it < 64; ++it) {
    int idx = tid + it * 128;          // 0..8191
    int kp = idx >> 4;
    int d4 = (idx & 15) << 2;
    const uint2 v = *(const uint2*)(Vp + (size_t)kp * D_ + d4);
    Vt[(d4 + 0) * LVT + kp] = (unsigned short)(v.x & 0xffffu);
    Vt[(d4 + 1) * LVT + kp] = (unsigned short)(v.x >> 16);
    Vt[(d4 + 2) * LVT + kp] = (unsigned short)(v.y & 0xffffu);
    Vt[(d4 + 3) * LVT + kp] = (unsigned short)(v.y >> 16);
  }

  // ---- Q fragments (this wave's 16 rows; scale already folded into Q) ----
  const int q0 = qt * 64 + wave * 16;
  FragBF qf[2];
  {
    const unsigned short* qr = Qp + (size_t)(q0 + l16) * D_;
#pragma unroll
    for (int hh = 0; hh < 2; ++hh) {
      int d0 = hh * 32;
#pragma unroll
      for (int j = 0; j < 8; ++j) {
        qf[hh].u[j]     = *(const unsigned int*)(qr + d0 + kb16 + 2 * j);
      }
    }
  }

  // ---- pass 1: scores = Q K^T ----
  for (int kt = 0; kt < L_ / 16; ++kt) {
    const unsigned short* kr = Kp + (size_t)(kt * 16 + l16) * D_;
    FragBF kf0, kf1;
#pragma unroll
    for (int j = 0; j < 8; ++j) {
      kf0.u[j] = *(const unsigned int*)(kr + kb16 + 2 * j);
      kf1.u[j] = *(const unsigned int*)(kr + 32 + kb16 + 2 * j);
    }
    AccF s;
#pragma unroll
    for (int i = 0; i < 8; ++i) s.f[i] = 0.0f;
    s.v = __builtin_amdgcn_wmma_f32_16x16x32_bf16(false, qf[0].v, false, kf0.v, (short)0, s.v, false, false);
    s.v = __builtin_amdgcn_wmma_f32_16x16x32_bf16(false, qf[1].v, false, kf1.v, (short)0, s.v, false, false);
#pragma unroll
    for (int i = 0; i < 8; ++i)
      sc[(wave * 16 + halo + i) * LSC + kt * 16 + l16] = s.f[i];
  }
  __syncthreads();

  // ---- pass 2: softmax per row (wave-parallel, shuffle reductions) ----
  for (int r = 0; r < 16; ++r) {
    float* row = &sc[(wave * 16 + r) * LSC];
    float m = -3.402823466e38f;
    for (int c = lane; c < L_; c += 32) m = fmaxf(m, row[c]);
    m = wred_max(m);
    float sum = 0.0f;
    for (int c = lane; c < L_; c += 32) {
      float e = __expf(row[c] - m);
      row[c] = e;
      sum += e;
    }
    sum = wred_sum(sum);
    float inv = 1.0f / sum;
    for (int c = lane; c < L_; c += 32) row[c] *= inv;
  }
  __syncthreads();

  // ---- pass 3: ctx = P @ V ----
  AccF o[4];
#pragma unroll
  for (int nt = 0; nt < 4; ++nt)
#pragma unroll
    for (int i = 0; i < 8; ++i) o[nt].f[i] = 0.0f;

  for (int kt = 0; kt < L_ / 32; ++kt) {
    FragBF pf;
    const float* pr = &sc[(wave * 16 + l16) * LSC + kt * 32];
#pragma unroll
    for (int j = 0; j < 8; ++j) {
      float2 a  = *(const float2*)(pr + kb16 + 2 * j);
      pf.u[j]     = pack2bf(a.x, a.y);
    }
#pragma unroll
    for (int nt = 0; nt < 4; ++nt) {
      FragBF vf;
      const unsigned short* vr = &Vt[(nt * 16 + l16) * LVT + kt * 32 + kb16];
#pragma unroll
      for (int j = 0; j < 8; ++j) vf.u[j] = *(const unsigned int*)(vr + 2 * j);
      o[nt].v = __builtin_amdgcn_wmma_f32_16x16x32_bf16(
          false, pf.v, false, vf.v, (short)0, o[nt].v, false, false);
    }
  }
#pragma unroll
  for (int nt = 0; nt < 4; ++nt) {
    int dh = nt * 16 + l16;
#pragma unroll
    for (int i = 0; i < 8; ++i) {
      int qr2 = q0 + halo + i;
      Op[(size_t)qr2 * D_ + dh] = f2bf(o[nt].f[i]);
    }
  }
}

// ---------------------------------------------------------------------------
// Host orchestration
// ---------------------------------------------------------------------------
extern "C" void kernel_launch(void* const* d_in, const int* in_sizes, int n_in,
                              void* d_out, int out_size, void* d_ws, size_t ws_size,
                              hipStream_t stream) {
  (void)in_sizes; (void)n_in; (void)out_size; (void)ws_size;
  const float* x_a  = (const float*)d_in[0];
  const float* x_b  = (const float*)d_in[1];
  const float* spk  = (const float*)d_in[2];
  const float* Wq   = (const float*)d_in[3];
  const float* bq   = (const float*)d_in[4];
  const float* Wk   = (const float*)d_in[5];
  const float* bk   = (const float*)d_in[6];
  const float* Wv   = (const float*)d_in[7];
  const float* bv   = (const float*)d_in[8];
  const float* Wo   = (const float*)d_in[9];
  const float* bo   = (const float*)d_in[10];
  const float* ln_g = (const float*)d_in[11];
  const float* ln_b = (const float*)d_in[12];
  const float* fg   = (const float*)d_in[13];
  const float* fb   = (const float*)d_in[14];
  const float* W1   = (const float*)d_in[15];
  const float* b1   = (const float*)d_in[16];
  const float* W2   = (const float*)d_in[17];
  const float* b2   = (const float*)d_in[18];

  char* ws = (char*)d_ws;
  size_t off = 0;
  auto take = [&](size_t bytes) -> char* {
    char* p = ws + off;
    off += (bytes + 255) & ~(size_t)255;
    return p;
  };
  const size_t nDD = (size_t)NL_ * D_ * D_;
  const size_t nDF = (size_t)NL_ * D_ * DFF_;
  const size_t nAct = (size_t)BL_ * D_;

  unsigned short* wqb = (unsigned short*)take(nDD * 2);
  unsigned short* wkb = (unsigned short*)take(nDD * 2);
  unsigned short* wvb = (unsigned short*)take(nDD * 2);
  unsigned short* wob = (unsigned short*)take(nDD * 2);
  unsigned short* w1b = (unsigned short*)take(nDF * 2);
  unsigned short* w2b = (unsigned short*)take(nDF * 2);
  unsigned short* xab   = (unsigned short*)take(nAct * 2);
  float*          xbf   = (float*)take(nAct * 4);
  float*          xbn   = (float*)take(nAct * 4);
  unsigned short* xbnb  = (unsigned short*)take(nAct * 2);
  unsigned short* qb    = (unsigned short*)take(nAct * 2);
  unsigned short* kb2   = (unsigned short*)take(nAct * 2);
  unsigned short* vb2   = (unsigned short*)take(nAct * 2);
  float*          outf  = (float*)take(nAct * 4);
  unsigned short* interb = (unsigned short*)take((size_t)BL_ * DFF_ * 2);
  unsigned short* ctxb = qb;     // alias: Q consumed before ctx written
  unsigned short* ffnb = xbnb;   // alias: xb_n bf16 consumed before ffn-LN

  cvt_bf16_kernel<<<4096, 256, 0, stream>>>(Wq, wqb, nDD);
  cvt_bf16_kernel<<<4096, 256, 0, stream>>>(Wk, wkb, nDD);
  cvt_bf16_kernel<<<4096, 256, 0, stream>>>(Wv, wvb, nDD);
  cvt_bf16_kernel<<<4096, 256, 0, stream>>>(Wo, wob, nDD);
  cvt_bf16_kernel<<<4096, 256, 0, stream>>>(W1, w1b, nDF);
  cvt_bf16_kernel<<<4096, 256, 0, stream>>>(W2, w2b, nDF);

  const int peBlocks = (int)(((size_t)BL_ * D_ + 255) / 256);
  add_pe_kernel<<<peBlocks, 256, 0, stream>>>(x_a, spk, nullptr, xab);
  add_pe_kernel<<<peBlocks, 256, 0, stream>>>(x_b, spk, xbf, xbnb);

  dim3 gD(D_ / 128, BL_ / 128);    // (8, 128)
  dim3 gF(DFF_ / 128, BL_ / 128);  // (32, 128)
  dim3 gA(L_ / 64, H_, B_);        // (8, 16, 32)
  const size_t attnSmem = (size_t)64 * LSC * 4 + (size_t)64 * LVT * 2;
  const float qscale = 0.125f;     // 1/sqrt(64)

  for (int i = 0; i < NL_; ++i) {
    const float* resid;
    const unsigned short* a_in;
    if (i == 0) { resid = xbf; a_in = xbnb; }
    else {
      layernorm_kernel<<<BL_, 256, 0, stream>>>(xbf, ln_g + (size_t)i * D_,
                                                ln_b + (size_t)i * D_, xbn, xbnb);
      resid = xbn; a_in = xbnb;
    }
    gemm_bf16_wmma<<<gD, 256, 0, stream>>>(a_in, wqb + (size_t)i * D_ * D_,
        bq + (size_t)i * D_, nullptr, nullptr, qb, BL_, D_, D_, qscale, 0);
    gemm_bf16_wmma<<<gD, 256, 0, stream>>>(xab, wkb + (size_t)i * D_ * D_,
        bk + (size_t)i * D_, nullptr, nullptr, kb2, BL_, D_, D_, 1.0f, 0);
    gemm_bf16_wmma<<<gD, 256, 0, stream>>>(xab, wvb + (size_t)i * D_ * D_,
        bv + (size_t)i * D_, nullptr, nullptr, vb2, BL_, D_, D_, 1.0f, 0);

    attention_wmma<<<gA, 128, attnSmem, stream>>>(qb, kb2, vb2, ctxb);

    gemm_bf16_wmma<<<gD, 256, 0, stream>>>(ctxb, wob + (size_t)i * D_ * D_,
        bo + (size_t)i * D_, resid, outf, nullptr, BL_, D_, D_, 1.0f, 0);

    layernorm_kernel<<<BL_, 256, 0, stream>>>(outf, fg + (size_t)i * D_,
                                              fb + (size_t)i * D_, nullptr, ffnb);
    gemm_bf16_wmma<<<gF, 256, 0, stream>>>(ffnb, w1b + (size_t)i * D_ * DFF_,
        b1 + (size_t)i * DFF_, nullptr, nullptr, interb, BL_, DFF_, D_, 1.0f, 1);

    float* xbOut = (i == NL_ - 1) ? (float*)d_out : xbf;
    gemm_bf16_wmma<<<gD, 256, 0, stream>>>(interb, w2b + (size_t)i * DFF_ * D_,
        b2 + (size_t)i * D_, outf, xbOut, nullptr, BL_, D_, DFF_, 1.0f, 0);
  }
}
